// MultiHeadAttention_56169582297160
// MI455X (gfx1250) — matrix operand order repaired and checked by reading, not compiled, not run on hardware
//
#include <hip/hip_runtime.h>
#include <stdint.h>

// Problem constants
#define BB 4
#define SS 1024
#define DM 512
#define NH 8
#define DK 64
#define MTOT (BB*SS)         // 4096 rows of the token-major GEMMs
#define LN_EPS 1e-3f

// ---------------- WMMA fragment machinery (gfx1250, wave32) ----------------
typedef __attribute__((ext_vector_type(16))) __bf16 v16bf;
typedef __attribute__((ext_vector_type(8)))  float  v8f;

union Frag16 { v16bf v; uint32_t d[8]; };
union FragC  { v8f   v; float    f[8]; };

__device__ __forceinline__ v8f wmma_bf16(v16bf a, v16bf b, v8f c) {
  // (neg_a, A, neg_b, B, c_mod, C, reuse_a, reuse_b)
  return __builtin_amdgcn_wmma_f32_16x16x32_bf16(false, a, false, b, (short)0, c, false, false);
}

// fp32 -> bf16 round-to-nearest-even
__device__ __forceinline__ unsigned short f2bf(float x) {
  union { float f; uint32_t u; } a; a.f = x;
  uint32_t r = a.u + 0x7FFFu + ((a.u >> 16) & 1u);
  return (unsigned short)(r >> 16);
}

// A fragment: 16x32 bf16 tile, row-major source, A points at tile origin.
// lane: m = lane&15, half = lane>>4 ; vgpr v holds K = (v>>2)*16 + half*8 + (v&3)*2 (+1)
__device__ __forceinline__ Frag16 load_frag_a(const unsigned short* __restrict__ A,
                                              int lda, int lane) {
  Frag16 f;
  const int m = lane & 15, hf = lane >> 4;
  const unsigned short* row = A + m * lda + hf * 8;
#pragma unroll
  for (int v = 0; v < 8; ++v) {
    const int k = ((v >> 2) << 4) + ((v & 3) << 1);
    f.d[v] = *(const uint32_t*)(row + k);
  }
  return f;
}

// B fragment: 32x16 bf16, where B[k][n] = W[n][k] with W row-major [N][K]
// (i.e. computes X * W^T). W points at tile origin (n0 row, k0 col).
// lane: n = lane&15, half = lane>>4 ; vgpr v holds K = (v>>2)*16 + half*8 + (v&3)*2 (+1)
__device__ __forceinline__ Frag16 load_frag_b(const unsigned short* __restrict__ W,
                                              int ldw, int lane) {
  Frag16 f;
  const int n = lane & 15, hf = lane >> 4;
  const unsigned short* row = W + n * ldw + hf * 8;
#pragma unroll
  for (int v = 0; v < 8; ++v) f.d[v] = *(const uint32_t*)(row + ((v >> 2) << 4) + ((v & 3) << 1));
  return f;
}

// ---------------- elementwise fp32 -> bf16 convert ----------------
__global__ void k_f2bf(const float* __restrict__ in, unsigned short* __restrict__ out, int n) {
  int i = blockIdx.x * 256 + threadIdx.x;
  if (i < n) out[i] = f2bf(in[i]);
}

// ---------------- fused QKV head projections ----------------
// qh[h][b][s][k], kh[h][b][s][k]  (row-major 64 per token)
// vhT[h][b][k][s]                 (transposed for the 2nd attention GEMM)
__global__ __launch_bounds__(256) void k_proj_qkv(
    const unsigned short* __restrict__ qb, const unsigned short* __restrict__ kb,
    const unsigned short* __restrict__ vb,
    const unsigned short* __restrict__ wq, const unsigned short* __restrict__ wk,
    const unsigned short* __restrict__ wv,
    unsigned short* __restrict__ qh, unsigned short* __restrict__ kh,
    unsigned short* __restrict__ vhT) {
  const int lane = threadIdx.x & 31;
  const int wid  = blockIdx.x * 8 + (threadIdx.x >> 5);   // 24576 tiles = 3 * 256 * 32
  const int mat  = wid >> 13;                             // 0:q 1:k 2:v
  const int t    = wid & 8191;
  const int mt = t >> 5, nt = t & 31;
  const int m0 = mt * 16, n0 = nt * 16;

  const unsigned short* A = (mat == 0) ? qb : (mat == 1) ? kb : vb;
  const unsigned short* W = (mat == 0) ? wq : (mat == 1) ? wk : wv;

  v8f c = {};
#pragma unroll 4
  for (int k0 = 0; k0 < DM; k0 += 32) {
    if (k0 + 32 < DM) {
      __builtin_prefetch(A + m0 * DM + k0 + 32);
      __builtin_prefetch(W + n0 * DM + k0 + 32);
    }
    Frag16 fa = load_frag_a(A + m0 * DM + k0, DM, lane);
    Frag16 fb = load_frag_b(W + n0 * DM + k0, DM, lane);
    c = wmma_bf16(fa.v, fb.v, c);
  }
  FragC cf; cf.v = c;

  const int hf  = lane >> 4;
  const int col = n0 + (lane & 15);
  const int h = col >> 6, kk = col & 63;
#pragma unroll
  for (int v = 0; v < 8; ++v) {
    const int m = m0 + v + 8 * hf;
    const int b = m >> 10, s = m & 1023;
    const unsigned short val = f2bf(cf.f[v]);
    if (mat == 2)      vhT[(((size_t)(h * BB + b) * DK + kk) * SS) + s] = val;
    else if (mat == 1) kh [(((size_t)(h * BB + b) * SS + s) * DK) + kk] = val;
    else               qh [(((size_t)(h * BB + b) * SS + s) * DK) + kk] = val;
  }
}

// ---------------- attention: scores + softmax + attn write + attn@V ----------------
#define PROBS_LD 1032   // padded probs row (ushorts)
#define KHS_LD   72     // padded K-slice row (ushorts): 36 dwords -> conflict-free frag reads
#define KHS_BYTES (SS * KHS_LD * 2)                 // 147,456
#define PROBS_OFF 0
#define PART_OFF  (16 * PROBS_LD * 2)               // 33,024 (within dead khs region)

__global__ __launch_bounds__(256) void k_attn(
    const unsigned short* __restrict__ qh, const unsigned short* __restrict__ kh,
    const unsigned short* __restrict__ vhT,
    float* __restrict__ attn_out, unsigned short* __restrict__ head_out) {
  extern __shared__ char smem[];
  // phase A: smem[0 .. KHS_BYTES) = khs[1024][KHS_LD] bf16 (K matrix staged via async DMA)
  // phase B (after 1st barrier): probs[16][PROBS_LD] bf16 + part[2][16*64] f32 alias khs
  unsigned short* khs   = (unsigned short*)smem;
  unsigned short* probs = (unsigned short*)(smem + PROBS_OFF);
  float*          part  = (float*)(smem + PART_OFF);
  __shared__ float redmax[8 * 16];
  __shared__ float redsum[8 * 16];

  const int tx = threadIdx.x, wave = tx >> 5, lane = tx & 31;
  const int hf = lane >> 4, nl = lane & 15;
  const int wg = blockIdx.x;            // H*B*64 workgroups
  const int mt = wg & 63, hb = wg >> 6; // hb = h*B + b
  const int m0 = mt * 16;

  const unsigned short* qbase = qh  + (size_t)hb * SS * DK;
  const unsigned short* kbase = kh  + (size_t)hb * SS * DK;
  const unsigned short* vtb   = vhT + (size_t)hb * DK * SS;

  // ---- async DMA: wave w stages K rows [w*128, w*128+128) into its LDS slice.
  // 128 rows * 8 segments of 16B = 1024 b128 transfers = 32 per lane.
#pragma unroll 4
  for (int it = 0; it < 32; ++it) {
    const int idx = it * 32 + lane;
    const int row = wave * 128 + (idx >> 3);
    const int seg = idx & 7;
    const unsigned short* gsrc = kbase + row * DK + seg * 8;
    const uint32_t ldst = (uint32_t)(uintptr_t)(khs + row * KHS_LD + seg * 8);
    asm volatile("global_load_async_to_lds_b128 %0, %1, off"
                 :: "v"(ldst), "v"(gsrc) : "memory");
  }

  // overlap: fetch the two Q A-fragments from global while the DMA runs
  Frag16 a0 = load_frag_a(qbase + m0 * DK + 0,  DK, lane);
  Frag16 a1 = load_frag_a(qbase + m0 * DK + 32, DK, lane);

  asm volatile("s_wait_asynccnt 0x0" ::: "memory");   // per-wave: own slice ready

  // ---- stage 1: scores stripe [16 x 1024], wave w owns contiguous n-tiles 8w..8w+7
  FragC sc[8];
#pragma unroll
  for (int i = 0; i < 8; ++i) {
    const int n0 = (wave * 8 + i) * 16;
    v8f c = {};
    Frag16 b0 = load_frag_b(khs + n0 * KHS_LD + 0,  KHS_LD, lane);
    c = wmma_bf16(a0.v, b0.v, c);
    Frag16 b1 = load_frag_b(khs + n0 * KHS_LD + 32, KHS_LD, lane);
    c = wmma_bf16(a1.v, b1.v, c);
    sc[i].v = c;
  }

  // ---- stage 2: softmax over the 1024 columns of each of the 16 rows
  const float scale = 0.044194173824159216f;  // 1/sqrt(D_MODEL) per reference
  float gmax[8], gsum[8];
#pragma unroll
  for (int v = 0; v < 8; ++v) {
    float pm = -3.4e38f;
#pragma unroll
    for (int i = 0; i < 8; ++i) { sc[i].f[v] *= scale; pm = fmaxf(pm, sc[i].f[v]); }
    pm = fmaxf(pm, __shfl_xor(pm, 1));
    pm = fmaxf(pm, __shfl_xor(pm, 2));
    pm = fmaxf(pm, __shfl_xor(pm, 4));
    pm = fmaxf(pm, __shfl_xor(pm, 8));
    if (nl == 0) redmax[wave * 16 + v + 8 * hf] = pm;
  }
  __syncthreads();   // also retires all khs reads -> khs region reusable after this
#pragma unroll
  for (int v = 0; v < 8; ++v) {
    const int r = v + 8 * hf;
    float gm = redmax[r];
#pragma unroll
    for (int w = 1; w < 8; ++w) gm = fmaxf(gm, redmax[w * 16 + r]);
    gmax[v] = gm;
  }
#pragma unroll
  for (int v = 0; v < 8; ++v) {
    float ps = 0.f;
#pragma unroll
    for (int i = 0; i < 8; ++i) {
      const float e = __expf(sc[i].f[v] - gmax[v]);
      sc[i].f[v] = e; ps += e;
    }
    ps += __shfl_xor(ps, 1);
    ps += __shfl_xor(ps, 2);
    ps += __shfl_xor(ps, 4);
    ps += __shfl_xor(ps, 8);
    if (nl == 0) redsum[wave * 16 + v + 8 * hf] = ps;
  }
  __syncthreads();
#pragma unroll
  for (int v = 0; v < 8; ++v) {
    const int r = v + 8 * hf;
    float gs = redsum[r];
#pragma unroll
    for (int w = 1; w < 8; ++w) gs += redsum[w * 16 + r];
    gsum[v] = 1.f / gs;
  }

  // ---- write attention probabilities (fp32 output tensor + bf16 LDS copy)
  float* arow = attn_out + ((size_t)hb * SS + m0) * SS;
#pragma unroll
  for (int i = 0; i < 8; ++i) {
    const int n = (wave * 8 + i) * 16 + nl;
#pragma unroll
    for (int v = 0; v < 8; ++v) {
      const int r = v + 8 * hf;
      const float p = sc[i].f[v] * gsum[v];
      arow[(size_t)r * SS + n] = p;
      probs[r * PROBS_LD + n]  = f2bf(p);
    }
  }
  __syncthreads();

  // ---- stage 3: out[16x64] = probs[16x1024] @ vh[1024x64]
  // wave -> (ntile = wave&3, khalf = wave>>2); fixed-order 2-partial reduction
  {
    const int ntile = wave & 3, khalf = wave >> 2;
    v8f c = {};
#pragma unroll
    for (int kk = 0; kk < 16; ++kk) {
      const int k0 = khalf * 512 + kk * 32;
      Frag16 fa = load_frag_a(probs + k0, PROBS_LD, lane);
      Frag16 fb = load_frag_b(vtb + (size_t)(ntile * 16) * SS + k0, SS, lane);
      c = wmma_bf16(fa.v, fb.v, c);
    }
    FragC cf; cf.v = c;
#pragma unroll
    for (int v = 0; v < 8; ++v)
      part[khalf * 1024 + (v + 8 * hf) * 64 + ntile * 16 + nl] = cf.f[v];
  }
  __syncthreads();

  // ---- head_out[b*S + s][h*64 + c]  (bf16, feeds the output projection GEMM)
  const int h = hb >> 2, b = hb & 3;
  for (int i = tx; i < 16 * 64; i += 256) {
    const int r = i >> 6, cc = i & 63;
    const float s = part[i] + part[1024 + i];
    head_out[((size_t)(b * SS) + m0 + r) * (NH * DK) + h * DK + cc] = f2bf(s);
  }
}

// ---------------- output projection + bias + residual ----------------
__global__ __launch_bounds__(256) void k_proj_out(
    const unsigned short* __restrict__ ho, const unsigned short* __restrict__ pw,
    const float* __restrict__ pb, const float* __restrict__ qres,
    float* __restrict__ z) {
  const int lane = threadIdx.x & 31;
  const int wid  = blockIdx.x * 8 + (threadIdx.x >> 5);   // 8192 tiles = 256 * 32
  const int mt = wid >> 5, nt = wid & 31;
  const int m0 = mt * 16, n0 = nt * 16;

  v8f c = {};
#pragma unroll 4
  for (int k0 = 0; k0 < DM; k0 += 32) {
    if (k0 + 32 < DM) {
      __builtin_prefetch(ho + m0 * DM + k0 + 32);
      __builtin_prefetch(pw + n0 * DM + k0 + 32);
    }
    Frag16 fa = load_frag_a(ho + m0 * DM + k0, DM, lane);
    Frag16 fb = load_frag_b(pw + n0 * DM + k0, DM, lane);
    c = wmma_bf16(fa.v, fb.v, c);
  }
  FragC cf; cf.v = c;

  const int hf = lane >> 4;
  const int col = n0 + (lane & 15);
  const float bias = pb[col];
#pragma unroll
  for (int v = 0; v < 8; ++v) {
    const int m = m0 + v + 8 * hf;
    z[(size_t)m * DM + col] = cf.f[v] + bias + qres[(size_t)m * DM + col];
  }
}

// ---------------- LayerNorm over the SEQUENCE axis (ref: axis=1, ddof=1) ----------------
__global__ __launch_bounds__(256) void k_ln(
    const float* __restrict__ z, const float* __restrict__ a2,
    const float* __restrict__ b2, float* __restrict__ out) {
  __shared__ float rs[8][32], rq[8][32], sc1[32], sc0[32];
  const int b = blockIdx.x >> 4;            // 16 d-chunks per batch (512/32)
  const int d = (blockIdx.x & 15) * 32 + (threadIdx.x & 31);
  const int sg = threadIdx.x >> 5;          // 8 sequence groups of 128
  const float* base = z + (size_t)b * SS * DM + d;

  float sum = 0.f, sq = 0.f;
  for (int s = sg * 128; s < sg * 128 + 128; ++s) {
    const float x = base[(size_t)s * DM];
    sum += x; sq += x * x;
  }
  rs[sg][threadIdx.x & 31] = sum;
  rq[sg][threadIdx.x & 31] = sq;
  __syncthreads();
  if (sg == 0) {
    float S1 = 0.f, S2 = 0.f;
#pragma unroll
    for (int w = 0; w < 8; ++w) { S1 += rs[w][threadIdx.x & 31]; S2 += rq[w][threadIdx.x & 31]; }
    const float mu  = S1 * (1.f / SS);
    const float var = (S2 - S1 * S1 * (1.f / SS)) * (1.f / (SS - 1));
    const float sig = sqrtf(fmaxf(var, 0.f));
    const float s1  = a2[d] / (sig + LN_EPS);
    sc1[threadIdx.x & 31] = s1;
    sc0[threadIdx.x & 31] = b2[d] - mu * s1;
  }
  __syncthreads();
  const float s1 = sc1[threadIdx.x & 31], s0 = sc0[threadIdx.x & 31];
  for (int s = sg * 128; s < sg * 128 + 128; ++s)
    out[((size_t)b * SS + s) * DM + d] = base[(size_t)s * DM] * s1 + s0;
}

// ---------------- host-side orchestration ----------------
extern "C" void kernel_launch(void* const* d_in, const int* in_sizes, int n_in,
                              void* d_out, int out_size, void* d_ws, size_t ws_size,
                              hipStream_t stream) {
  const float* q   = (const float*)d_in[0];
  const float* k   = (const float*)d_in[1];
  const float* v   = (const float*)d_in[2];
  const float* Wq  = (const float*)d_in[3];
  const float* Wk  = (const float*)d_in[4];
  const float* Wv  = (const float*)d_in[5];
  const float* pw  = (const float*)d_in[6];
  const float* pb  = (const float*)d_in[7];
  const float* a2  = (const float*)d_in[8];
  const float* b2  = (const float*)d_in[9];

  // workspace layout, ~38 MB total
  char* ws = (char*)d_ws;
  const size_t NTOK = (size_t)MTOT * DM;        // 2,097,152
  const size_t NW   = (size_t)DM * DM;          //   262,144
  unsigned short* qb  = (unsigned short*)(ws);
  unsigned short* kb  = qb  + NTOK;
  unsigned short* vb  = kb  + NTOK;
  unsigned short* wqb = vb  + NTOK;
  unsigned short* wkb = wqb + NW;
  unsigned short* wvb = wkb + NW;
  unsigned short* pwb = wvb + NW;
  unsigned short* qh  = pwb + NW;               // [H][B][S][64]
  unsigned short* kh  = qh  + NTOK;
  unsigned short* vhT = kh  + NTOK;             // [H][B][64][S]
  unsigned short* ho  = vhT + NTOK;             // [B*S][H*64]
  float*          z   = (float*)(ho + NTOK);    // [B*S][D_MODEL]

  float* ln_out   = (float*)d_out;
  float* attn_out = ln_out + NTOK;              // [H*B][S][S]

  // fp32 -> bf16 converts
  k_f2bf<<<(int)((NTOK + 255) / 256), 256, 0, stream>>>(q,  qb,  (int)NTOK);
  k_f2bf<<<(int)((NTOK + 255) / 256), 256, 0, stream>>>(k,  kb,  (int)NTOK);
  k_f2bf<<<(int)((NTOK + 255) / 256), 256, 0, stream>>>(v,  vb,  (int)NTOK);
  k_f2bf<<<(int)((NW   + 255) / 256), 256, 0, stream>>>(Wq, wqb, (int)NW);
  k_f2bf<<<(int)((NW   + 255) / 256), 256, 0, stream>>>(Wk, wkb, (int)NW);
  k_f2bf<<<(int)((NW   + 255) / 256), 256, 0, stream>>>(Wv, wvb, (int)NW);
  k_f2bf<<<(int)((NW   + 255) / 256), 256, 0, stream>>>(pw, pwb, (int)NW);

  // per-head projections: 3 * (256 m-tiles * 32 n-tiles) tiles, 8 waves/block
  k_proj_qkv<<<3072, 256, 0, stream>>>(qb, kb, vb, wqb, wkb, wvb, qh, kh, vhT);

  // attention: H*B*(S/16) = 2048 workgroups, 144KB dynamic LDS (K stage + aliased probs)
  k_attn<<<2048, 256, KHS_BYTES, stream>>>(qh, kh, vhT, attn_out, ho);

  // output projection + residual: 8192 tiles / 8 waves
  k_proj_out<<<1024, 256, 0, stream>>>(ho, pwb, pb, q, z);

  // layernorm over sequence axis: B * 16 d-chunks
  k_ln<<<64, 256, 0, stream>>>(z, a2, b2, ln_out);
}
